// SLSTMBlock_83777632076211
// MI455X (gfx1250) — matrix operand order repaired and checked by reading, not compiled, not run on hardware
//
#include <hip/hip_runtime.h>
#include <hip/hip_bf16.h>

// Problem constants (from reference)
#define Bsz   2048
#define Hdim  4096
#define NHead 4
#define HSd   1024   // head size
// conv taps K = 4 (3 state taps + current xn)

typedef __bf16 bf16_t;
typedef __attribute__((ext_vector_type(16))) __bf16 v16bf;
typedef __attribute__((ext_vector_type(8)))  __bf16 v8bf;
typedef __attribute__((ext_vector_type(4)))  __bf16 v4bf;
typedef __attribute__((ext_vector_type(8)))  float  v8f;

__device__ __forceinline__ v8bf ldv8(const bf16_t* p) {
  return *reinterpret_cast<const v8bf*>(p);
}

__device__ __forceinline__ v16bf frag_cat(v8bf lo, v8bf hi) {
  v16bf r;
#pragma unroll
  for (int i = 0; i < 8; ++i) { r[i] = lo[i]; r[i + 8] = hi[i]; }
  return r;
}

// Async global -> LDS copy of 16 bytes per lane (gfx1250, tracked by ASYNCcnt).
// VDST VGPR carries the per-lane LDS byte address (low 32 bits of the generic
// shared pointer == LDS offset per the flat-aperture rule).
__device__ __forceinline__ void async_cp16(unsigned lds_addr, const bf16_t* gptr) {
  asm volatile("global_load_async_to_lds_b128 %0, %1, off"
               :: "v"(lds_addr), "v"(gptr) : "memory");
}

// ---------------------------------------------------------------------------
// Kernel 1: LayerNorm + causal conv(K=4) + SiLU; emit new_conv_state and the
// bf16 GEMM operands (xn, x_act, recurrent) into workspace.
// ---------------------------------------------------------------------------
__global__ __launch_bounds__(256)
void slstm_prologue(const float* __restrict__ x,
                    const float* __restrict__ conv_state,   // (B,3,H)
                    const float* __restrict__ recurrent,    // (B,H)
                    const float* __restrict__ ln_w,         // (H)
                    const float* __restrict__ conv_w,       // (H,4)
                    const float* __restrict__ conv_b,       // (H)
                    float*  __restrict__ ncs_out,           // (B,3,H)
                    bf16_t* __restrict__ xnb,
                    bf16_t* __restrict__ xab,
                    bf16_t* __restrict__ rb)
{
  const int b   = blockIdx.x;
  const int tid = threadIdx.x;
  const float* xr = x + (size_t)b * Hdim;

  __shared__ float s1[256], s2[256];
  float sum = 0.f, sq = 0.f;
  for (int h = tid; h < Hdim; h += 256) {
    float v = xr[h];
    sum += v; sq += v * v;
  }
  s1[tid] = sum; s2[tid] = sq;
  __syncthreads();
  for (int s = 128; s > 0; s >>= 1) {
    if (tid < s) { s1[tid] += s1[tid + s]; s2[tid] += s2[tid + s]; }
    __syncthreads();
  }
  const float mean = s1[0] * (1.f / Hdim);
  const float var  = s2[0] * (1.f / Hdim) - mean * mean;
  const float rstd = rsqrtf(var + 1e-5f);

  for (int h = tid; h < Hdim; h += 256) {
    const float xv = xr[h];
    const float xn = (xv - mean) * rstd * ln_w[h];
    const float c0 = conv_state[((size_t)b * 3 + 0) * Hdim + h];
    const float c1 = conv_state[((size_t)b * 3 + 1) * Hdim + h];
    const float c2 = conv_state[((size_t)b * 3 + 2) * Hdim + h];
    const float* cw = conv_w + (size_t)h * 4;
    const float xc = c0 * cw[0] + c1 * cw[1] + c2 * cw[2] + xn * cw[3] + conv_b[h];
    const float xa = xc / (1.f + __expf(-xc));   // SiLU

    ncs_out[((size_t)b * 3 + 0) * Hdim + h] = c1;
    ncs_out[((size_t)b * 3 + 1) * Hdim + h] = c2;
    ncs_out[((size_t)b * 3 + 2) * Hdim + h] = xn;

    const size_t idx = (size_t)b * Hdim + h;
    xnb[idx] = (bf16_t)xn;
    xab[idx] = (bf16_t)xa;
    rb[idx]  = (bf16_t)recurrent[idx];
  }
}

// ---------------------------------------------------------------------------
// Kernel 2: convert the 8 gate weight tensors (NH,HS,HS) f32 -> bf16.
// Order: Wi_in, Wf_in, Wz_in, Wo_in, Wi_st, Wf_st, Wz_st, Wo_st.
// ---------------------------------------------------------------------------
__global__ __launch_bounds__(256)
void slstm_cvtw(const float* __restrict__ w0, const float* __restrict__ w1,
                const float* __restrict__ w2, const float* __restrict__ w3,
                const float* __restrict__ w4, const float* __restrict__ w5,
                const float* __restrict__ w6, const float* __restrict__ w7,
                bf16_t* __restrict__ wb)
{
  const float* srcs[8] = { w0, w1, w2, w3, w4, w5, w6, w7 };
  const int widx = blockIdx.y;
  const size_t WSZ = (size_t)NHead * HSd * HSd;
  const size_t i = ((size_t)blockIdx.x * 256 + threadIdx.x) * 4;
  const float4 v = *reinterpret_cast<const float4*>(srcs[widx] + i);
  v4bf o;
  o[0] = (bf16_t)v.x; o[1] = (bf16_t)v.y; o[2] = (bf16_t)v.z; o[3] = (bf16_t)v.w;
  *reinterpret_cast<v4bf*>(wb + widx * WSZ + i) = o;
}

// ---------------------------------------------------------------------------
// Kernel 3: fused 4-gate GEMM (bf16 WMMA, f32 accum) + sLSTM cell epilogue.
// Grid: (N/64, M/64, NH). Block: 256 threads = 8 waves (4 M-strips x 2 N-pairs).
// B (weight) tiles are staged global->LDS with ASYNC copies, double buffered:
//   per K-chunk (32): 4 gates x 64 cols x 32 K bf16 = 16 KB, staged once per WG
//   and consumed by all 4 M-strip waves from LDS (ds_load_b128), removing the
//   4x redundant global B traffic. A fragments load directly (contiguous in K).
// K is flattened to 64 chunks: chunks 0..31 = input GEMMs (A = x_act / xn),
// chunks 32..63 = state GEMMs (A = recurrent), weights *_in then *_st.
// ---------------------------------------------------------------------------
__global__ __launch_bounds__(256)
void slstm_gemm(const bf16_t* __restrict__ xab, const bf16_t* __restrict__ xnb,
                const bf16_t* __restrict__ rb,  const bf16_t* __restrict__ wb,
                const float* __restrict__ bi,  const float* __restrict__ bfv,
                const float* __restrict__ bz,  const float* __restrict__ bo,
                const float* __restrict__ cell_state,
                const float* __restrict__ norm_state,
                const float* __restrict__ max_state,
                float* __restrict__ y_new,   float* __restrict__ cell_new,
                float* __restrict__ norm_new, float* __restrict__ max_new)
{
  __shared__ __align__(16) bf16_t smemB[2][4 * 64 * 32];  // [buf][gate*64+col][k] 2x16KB

  const int head = blockIdx.z;
  const int m0   = blockIdx.y * 64;
  const int n0   = blockIdx.x * 64;
  const int tid  = threadIdx.x;
  const int lane = tid & 31;
  const int wave = tid >> 5;
  const int wm   = wave & 3;
  const int wn2  = wave >> 2;
  const int l16  = lane & 15;
  const int half = lane >> 4;

  // staging role: thread t copies the 32-K row of (gate sg, col sc)
  const int sg = tid >> 6;
  const int sc = tid & 63;

  const int mrow = m0 + wm * 16 + l16;
  const bf16_t* a_xa = xab + (size_t)mrow * Hdim + head * HSd + 16 * half;
  const bf16_t* a_xn = xnb + (size_t)mrow * Hdim + head * HSd + 16 * half;
  const bf16_t* a_r  = rb  + (size_t)mrow * Hdim + head * HSd + 16 * half;

  // staging source base (k added per chunk); weight widx = p*4 + sg
  const size_t WQ = (size_t)HSd * HSd;
  const bf16_t* wsrc0 = wb + ((size_t)sg * NHead + head) * WQ + (size_t)(n0 + sc) * HSd;
  const unsigned ldsrow = (unsigned)(uintptr_t)(&smemB[0][0]) + (unsigned)((sg * 64 + sc) * 32) * 2u;
  const unsigned ldsbuf = (unsigned)(4 * 64 * 32 * 2);  // byte stride between buffers

  v8f acc[4][2];
#pragma unroll
  for (int g = 0; g < 4; ++g)
#pragma unroll
    for (int j = 0; j < 2; ++j) { v8f z = {}; acc[g][j] = z; }

  int cl[2];
  cl[0] = (wn2 * 2 + 0) * 16 + l16;
  cl[1] = (wn2 * 2 + 1) * 16 + l16;

  // ---- stage chunk 0 into buffer 0
  {
    const bf16_t* src = wsrc0;              // p=0, kk=0
    async_cp16(ldsrow,      src);
    async_cp16(ldsrow + 16, src + 8);
    async_cp16(ldsrow + 32, src + 16);
    async_cp16(ldsrow + 48, src + 24);
  }

  for (int c = 0; c < 64; ++c) {
    const int buf = c & 1;

    // ---- stage next chunk into the other buffer
    if (c + 1 < 64) {
      const int p  = (c + 1) >> 5;
      const int kk = ((c + 1) & 31) * 32;
      const bf16_t* src = wsrc0 + (size_t)(p * 4) * NHead * WQ + kk;
      const unsigned dst = ldsrow + (buf ? 0u : ldsbuf);
      async_cp16(dst,      src);
      async_cp16(dst + 16, src + 8);
      async_cp16(dst + 32, src + 16);
      async_cp16(dst + 48, src + 24);
      asm volatile("s_wait_asynccnt 4" ::: "memory");  // chunk c fully landed
    } else {
      asm volatile("s_wait_asynccnt 0" ::: "memory");
    }
    __syncthreads();   // all waves' stores for chunk c visible; restaging above has no barrier after chunk c-1 reads

    // ---- A fragments (direct, contiguous in K)
    const int p  = c >> 5;
    const int kk = (c & 31) * 32;
    const bf16_t* aIF = (p ? a_r : a_xa) + kk;
    const bf16_t* aZO = (p ? a_r : a_xn) + kk;
    __builtin_prefetch(aIF + 256, 0, 1);
    const v16bf aif = frag_cat(ldv8(aIF), ldv8(aIF + 8));
    const v16bf azo = frag_cat(ldv8(aZO), ldv8(aZO + 8));

    // ---- B fragments from LDS + WMMA
    const bf16_t* bb = &smemB[buf][0] + half * 16;
#pragma unroll
    for (int j = 0; j < 2; ++j) {
      const bf16_t* r0 = bb + (0 * 64 + cl[j]) * 32;
      v16bf bI = frag_cat(ldv8(r0), ldv8(r0 + 8));
      acc[0][j] = __builtin_amdgcn_wmma_f32_16x16x32_bf16(
          false, aif, false, bI, (short)0, acc[0][j], false, false);
      const bf16_t* r1 = bb + (1 * 64 + cl[j]) * 32;
      v16bf bF = frag_cat(ldv8(r1), ldv8(r1 + 8));
      acc[1][j] = __builtin_amdgcn_wmma_f32_16x16x32_bf16(
          false, aif, false, bF, (short)0, acc[1][j], false, false);
      const bf16_t* r2 = bb + (2 * 64 + cl[j]) * 32;
      v16bf bZ = frag_cat(ldv8(r2), ldv8(r2 + 8));
      acc[2][j] = __builtin_amdgcn_wmma_f32_16x16x32_bf16(
          false, azo, false, bZ, (short)0, acc[2][j], false, false);
      const bf16_t* r3 = bb + (3 * 64 + cl[j]) * 32;
      v16bf bO = frag_cat(ldv8(r3), ldv8(r3 + 8));
      acc[3][j] = __builtin_amdgcn_wmma_f32_16x16x32_bf16(
          false, azo, false, bO, (short)0, acc[3][j], false, false);
    }
  }

  // --- sLSTM cell epilogue: C/D layout => element e: M = e + 8*half, N = lane%16
#pragma unroll
  for (int j = 0; j < 2; ++j) {
    const int hglob = head * HSd + cl[j] + n0;
    const float vbi = bi[hglob], vbf = bfv[hglob];
    const float vbz = bz[hglob], vbo = bo[hglob];
#pragma unroll
    for (int e = 0; e < 8; ++e) {
      const int brow = m0 + wm * 16 + e + 8 * half;
      const size_t idx = (size_t)brow * Hdim + hglob;
      const float iv = acc[0][j][e] + vbi;
      const float fv = acc[1][j][e] + vbf;
      const float zv = acc[2][j][e] + vbz;
      const float ov = acc[3][j][e] + vbo;
      const float ms = max_state[idx];
      const float cs = cell_state[idx];
      const float ns = norm_state[idx];
      // log_sigmoid(f) = min(f,0) - log1p(exp(-|f|))
      const float lsig = fminf(fv, 0.f) - log1pf(__expf(-fabsf(fv)));
      const float lfm  = ms + lsig;
      const float mx   = fmaxf(iv, lfm);
      const float ig   = __expf(iv - mx);
      const float fg   = __expf(lfm - mx);
      const float cn   = fg * cs + ig * tanhf(zv);
      const float nn   = fg * ns + ig;
      const float yn   = cn / ((nn + 1e-6f) * (1.f + __expf(-ov)));  // sigmoid(o)*cn/(nn+eps)
      y_new[idx]    = yn;
      cell_new[idx] = cn;
      norm_new[idx] = nn;
      max_new[idx]  = mx;
    }
  }
}

// ---------------------------------------------------------------------------
// Kernel 4: GroupNorm(y_new) * gn_w + gn_b + skip(x)  -> out.
// ---------------------------------------------------------------------------
__global__ __launch_bounds__(256)
void slstm_gnorm(const float* __restrict__ y, const float* __restrict__ x,
                 const float* __restrict__ gw, const float* __restrict__ gb,
                 float* __restrict__ out)
{
  const int bg = blockIdx.x;
  const int b  = bg >> 2;          // / NHead
  const int g  = bg & 3;           // % NHead
  const int tid = threadIdx.x;
  const float* yr = y + (size_t)b * Hdim + g * HSd;

  float v0[4];
  float sum = 0.f, sq = 0.f;
#pragma unroll
  for (int i = 0; i < 4; ++i) {
    const float v = yr[tid + i * 256];
    v0[i] = v; sum += v; sq += v * v;
  }
  __shared__ float s1[256], s2[256];
  s1[tid] = sum; s2[tid] = sq;
  __syncthreads();
  for (int s = 128; s > 0; s >>= 1) {
    if (tid < s) { s1[tid] += s1[tid + s]; s2[tid] += s2[tid + s]; }
    __syncthreads();
  }
  const float mean = s1[0] * (1.f / HSd);
  const float var  = s2[0] * (1.f / HSd) - mean * mean;
  const float rstd = rsqrtf(var + 1e-5f);
#pragma unroll
  for (int i = 0; i < 4; ++i) {
    const int hs = tid + i * 256;
    const int h  = g * HSd + hs;
    const size_t idx = (size_t)b * Hdim + h;
    out[idx] = (v0[i] - mean) * rstd * gw[h] + gb[h] + x[idx];
  }
}

// ---------------------------------------------------------------------------
extern "C" void kernel_launch(void* const* d_in, const int* in_sizes, int n_in,
                              void* d_out, int out_size, void* d_ws, size_t ws_size,
                              hipStream_t stream)
{
  (void)in_sizes; (void)n_in; (void)out_size; (void)ws_size;

  const float* x          = (const float*)d_in[0];
  const float* conv_state = (const float*)d_in[1];
  const float* recurrent  = (const float*)d_in[2];
  const float* cell_state = (const float*)d_in[3];
  const float* norm_state = (const float*)d_in[4];
  const float* max_state  = (const float*)d_in[5];
  const float* ln_w   = (const float*)d_in[6];
  const float* conv_w = (const float*)d_in[7];
  const float* conv_b = (const float*)d_in[8];
  const float* Wi_in  = (const float*)d_in[9];
  const float* Wf_in  = (const float*)d_in[10];
  const float* Wz_in  = (const float*)d_in[11];
  const float* Wo_in  = (const float*)d_in[12];
  const float* Wi_st  = (const float*)d_in[13];
  const float* Wf_st  = (const float*)d_in[14];
  const float* Wz_st  = (const float*)d_in[15];
  const float* Wo_st  = (const float*)d_in[16];
  const float* bi     = (const float*)d_in[17];
  const float* bf_    = (const float*)d_in[18];
  const float* bz     = (const float*)d_in[19];
  const float* bo     = (const float*)d_in[20];
  const float* gn_w   = (const float*)d_in[21];
  const float* gn_b   = (const float*)d_in[22];

  const size_t BH = (size_t)Bsz * Hdim;
  float* out      = (float*)d_out;        // (B,H)
  float* ncs      = out + BH;             // (B,3,H) new_conv_state
  float* y_new    = out + 4 * BH;         // (B,H)
  float* cell_new = out + 5 * BH;
  float* norm_new = out + 6 * BH;
  float* max_new  = out + 7 * BH;

  // Workspace: bf16 xn, x_act, recurrent (16 MB each) + bf16 weights (64 MB)
  bf16_t* xnb = (bf16_t*)d_ws;
  bf16_t* xab = xnb + BH;
  bf16_t* rbw = xab + BH;
  bf16_t* wb  = rbw + BH;

  slstm_prologue<<<Bsz, 256, 0, stream>>>(x, conv_state, recurrent,
                                          ln_w, conv_w, conv_b,
                                          ncs, xnb, xab, rbw);

  dim3 gcv((NHead * HSd * HSd) / (256 * 4), 8);
  slstm_cvtw<<<gcv, 256, 0, stream>>>(Wi_in, Wf_in, Wz_in, Wo_in,
                                      Wi_st, Wf_st, Wz_st, Wo_st, wb);

  dim3 gmm(HSd / 64, Bsz / 64, NHead);
  slstm_gemm<<<gmm, 256, 0, stream>>>(xab, xnb, rbw, wb,
                                      bi, bf_, bz, bo,
                                      cell_state, norm_state, max_state,
                                      y_new, cell_new, norm_new, max_new);

  slstm_gnorm<<<Bsz * NHead, 256, 0, stream>>>(y_new, x, gn_w, gn_b, out);
}
